// CausalSelfAttention_51488067945166
// MI455X (gfx1250) — matrix operand order repaired and checked by reading, not compiled, not run on hardware
//
#include <hip/hip_runtime.h>

// ---------------------------------------------------------------------------
// Causal self-attention for MI455X (gfx1250), bf16 WMMA everywhere.
// B=2, T=2048, D=1024, H=16, d_head=64.
// Workspace: Q,K,V [B,H,T,64] bf16 (24MB) + Y [B,T,1024] bf16 (8MB) -> 32MB,
// fully resident in the 192MB L2.
// GEMMs: wave tile 16x64 (4 WMMAs / K-step), weight panel staged into LDS by
// the Tensor Data Mover (tensor_load_to_lds + s_wait_tensorcnt).
// ---------------------------------------------------------------------------

typedef __bf16 bf16_t;
typedef __attribute__((ext_vector_type(16))) __bf16 v16bf;
typedef __attribute__((ext_vector_type(8)))  __bf16 v8bf;
typedef __attribute__((ext_vector_type(8)))  float  v8f;
typedef __attribute__((ext_vector_type(4)))  float  v4f;
typedef unsigned int u32x4 __attribute__((ext_vector_type(4)));
typedef int          i32x8 __attribute__((ext_vector_type(8)));
typedef int          i32x4 __attribute__((ext_vector_type(4)));

#define BB  2
#define TT  2048
#define DD  1024
#define NH  16
#define DH  64

// ---------------------------------------------------------------------------
// TDM: async 2D tile load (rows x cols of f32) from global to LDS.
// D# layout per CDNA5 ISA ch.10.8 / CDNA5 ISA.
// Descriptor values must be wave-uniform. EXEC is ignored by TDM.
// This toolchain uses the 6-arg builtin (g0, g1, g2, g3, g4, cpol).
// ---------------------------------------------------------------------------
__device__ __forceinline__ void tdm_load_tile_f32(unsigned lds_byte_off,
                                                  const float* gsrc,
                                                  unsigned tensor_w,   // row len (elems)
                                                  unsigned tensor_h,   // rows (elems)
                                                  unsigned tile_w,     // tile cols
                                                  unsigned tile_h,     // tile rows
                                                  unsigned row_stride) // elems
{
    unsigned long long ga = (unsigned long long)(const void*)gsrc;
    u32x4 g0;
    g0[0] = 1u;                                          // count=1 (valid user D#)
    g0[1] = lds_byte_off;                                // lds_addr
    g0[2] = (unsigned)(ga & 0xFFFFFFFFull);              // global_addr[31:0]
    g0[3] = (unsigned)((ga >> 32) & 0x01FFFFFFull)       // global_addr[56:32]
          | (2u << 30);                                  // type=2 ("image")
    i32x8 g1 = {0, 0, 0, 0, 0, 0, 0, 0};
    g1[0] = (int)(2u << 16);                             // data_size=2 (4 bytes), mask=0
    g1[1] = (int)((tensor_w & 0xFFFFu) << 16);           // tensor_dim0[15:0] @ bits 63:48
    g1[2] = (int)((tensor_w >> 16) | ((tensor_h & 0xFFFFu) << 16)); // td0 hi | td1 lo
    g1[3] = (int)((tensor_h >> 16) | ((tile_w & 0xFFFFu) << 16));   // td1 hi | tile_dim0
    g1[4] = (int)(tile_h & 0xFFFFu);                     // tile_dim1 (tile_dim2=0)
    g1[5] = (int)row_stride;                             // tensor_dim0_stride[31:0]
    g1[6] = 0;                                           // stride hi | dim1_stride lo
    g1[7] = 0;
    i32x4 gz4 = {0, 0, 0, 0};                            // groups 2/3 unused (2D tensor)
    i32x8 gz8 = {0, 0, 0, 0, 0, 0, 0, 0};                // trailing group (unused)
    __builtin_amdgcn_tensor_load_to_lds(g0, g1, gz4, gz4, gz8, 0);
}

__device__ __forceinline__ unsigned lds_addr_of(const void* p) {
    // flat LDS addresses carry the LDS byte offset in addr[31:0]
    return (unsigned)(unsigned long long)p;
}

// ---------------------------------------------------------------------------
// Kernel 1: QKV = X @ Wqkv  (M=4096, N=3072, K=1024), output bf16 Q/K/V in
// [B,H,T,d] layout. Q pre-scaled by d_head^-0.5 = 0.125.
// Block = 8 waves sharing one 32x64 f32 W panel (TDM-staged in LDS); each
// wave owns a 16-row X tile and computes a 16x64 output tile (4 WMMAs/step).
// ---------------------------------------------------------------------------
__global__ __launch_bounds__(256)
void qkv_gemm_kernel(const float* __restrict__ x, const float* __restrict__ w,
                     bf16_t* __restrict__ Qw, bf16_t* __restrict__ Kw,
                     bf16_t* __restrict__ Vw) {
    __shared__ float lds_wf[32 * 64];                    // 8KB staged W panel

    const int lane = threadIdx.x & 31;
    const int wv   = threadIdx.x >> 5;
    const int l16  = lane & 15;
    const int hi   = lane >> 4;
    const int ntg  = blockIdx.x;              // 0..47  (64-col N group)
    const int mt   = blockIdx.y * 8 + wv;     // 0..255 (16-row M tile)

    const float* xrow = x + (size_t)(mt * 16 + l16) * DD;   // A: row = lane%16
    const unsigned lds_off = lds_addr_of(lds_wf);

    v8f c[4] = {};
    for (int kk = 0; kk < DD; kk += 32) {
        __syncthreads();                       // all readers done with panel
        if (wv == 0) {
            tdm_load_tile_f32(lds_off, w + (size_t)kk * (3 * DD) + ntg * 64,
                              3 * DD, DD, 64, 32, 3 * DD);
            __builtin_amdgcn_s_wait_tensorcnt(0);
        }
        if (kk + 32 < DD) {
            int kloc = threadIdx.x >> 3, n = (threadIdx.x & 7) * 8;
            __builtin_prefetch(&w[(size_t)(kk + 32 + kloc) * (3 * DD) + ntg * 64 + n], 0, 1);
        }
        __syncthreads();                       // panel visible to all waves

        // A fragment: 16-bit A 16x32 layout: lane l16=row; K runs
        // [hi*8 .. +7] and [16+hi*8 .. +7] (contiguous f32 b128 loads)
        v4f r0 = *(const v4f*)(xrow + kk + hi * 8);
        v4f r1 = *(const v4f*)(xrow + kk + hi * 8 + 4);
        v4f r2 = *(const v4f*)(xrow + kk + 16 + hi * 8);
        v4f r3 = *(const v4f*)(xrow + kk + 16 + hi * 8 + 4);
        v16bf a;
#pragma unroll
        for (int i = 0; i < 4; ++i) {
            a[i]      = (bf16_t)r0[i];
            a[4 + i]  = (bf16_t)r1[i];
            a[8 + i]  = (bf16_t)r2[i];
            a[12 + i] = (bf16_t)r3[i];
        }
        // B fragments: 32x16, col = lane%16, K = hi*8 + e + (e&8)
#pragma unroll
        for (int nt4 = 0; nt4 < 4; ++nt4) {
            v16bf bfrag;
#pragma unroll
            for (int e = 0; e < 16; ++e)
                bfrag[e] = (bf16_t)lds_wf[(hi * 8 + e + (e & 8)) * 64 + nt4 * 16 + l16];
            c[nt4] = __builtin_amdgcn_wmma_f32_16x16x32_bf16(false, a, false, bfrag,
                                                             (short)0, c[nt4],
                                                             false, false);
        }
    }

    // route to Q/K/V: n group of 64 is aligned, so s and h are block-uniform
    const int nbase = ntg * 64;
    const int s = nbase >> 10;                 // 0=Q 1=K 2=V
    const int h = (nbase >> 6) & 15;
    bf16_t* dst = (s == 0) ? Qw : (s == 1) ? Kw : Vw;
    const float qscale = (s == 0) ? 0.125f : 1.0f;   // d_head^-0.5
#pragma unroll
    for (int r = 0; r < 8; ++r) {
        int tok = mt * 16 + hi * 8 + r;
        int b = tok >> 11, t = tok & (TT - 1);
        bf16_t* drow = dst + (((size_t)b * NH + h) * TT + t) * DH;
#pragma unroll
        for (int nt4 = 0; nt4 < 4; ++nt4)
            drow[nt4 * 16 + l16] = (bf16_t)(c[nt4][r] * qscale);
    }
}

// ---------------------------------------------------------------------------
// Kernel 2: flash attention. One wave per (b,h, 16-query tile), key blocks
// of 32. Online softmax on the C-fragment layout; P re-shaped C->A layout
// through per-wave LDS; P@V via 4 bf16 WMMAs into a 16x64 accumulator.
// ---------------------------------------------------------------------------
__global__ __launch_bounds__(256)
void attn_kernel(const bf16_t* __restrict__ Qw, const bf16_t* __restrict__ Kw,
                 const bf16_t* __restrict__ Vw, bf16_t* __restrict__ Yw) {
    __shared__ bf16_t lds_p[8][16 * 32];   // per-wave P staging (1KB each)

    const int lane = threadIdx.x & 31;
    const int wv   = threadIdx.x >> 5;
    const int l16  = lane & 15;
    const int hi   = lane >> 4;
    const int tile = blockIdx.x * 8 + wv;  // 0..4095
    const int qt   = tile & 127;
    const int bh   = tile >> 7;            // 0..31
    const int q0   = qt * 16;

    const bf16_t* Qbh = Qw + (size_t)bh * TT * DH;
    const bf16_t* Kbh = Kw + (size_t)bh * TT * DH;
    const bf16_t* Vbh = Vw + (size_t)bh * TT * DH;

    // Q A-fragments for d-slices [0,32) and [32,64)
    const bf16_t* qrow = Qbh + (size_t)(q0 + l16) * DH;
    v16bf qa[2];
#pragma unroll
    for (int t2 = 0; t2 < 2; ++t2) {
        v8bf r0 = *(const v8bf*)(qrow + t2 * 32 + hi * 16);
        v8bf r1 = *(const v8bf*)(qrow + t2 * 32 + hi * 16 + 8);
#pragma unroll
        for (int i = 0; i < 8; ++i) { qa[t2][i] = r0[i]; qa[t2][8 + i] = r1[i]; }
    }

    float m[8], lsum[8];
#pragma unroll
    for (int r = 0; r < 8; ++r) { m[r] = -1e30f; lsum[r] = 0.0f; }
    v8f acc[4] = {};

    bf16_t* P = &lds_p[wv][0];
    const int nkb = (q0 + 15) / 32 + 1;

    for (int kb = 0; kb < nkb; ++kb) {
        const int k0 = kb * 32;
        // ---- S = Q @ K^T : two 16x16 tiles (keys k0..+15, k0+16..+31)
        v8f s0 = {}, s1 = {};
#pragma unroll
        for (int t2 = 0; t2 < 2; ++t2) {
            // B frag: B[k][n] = K[k0 + n][t2*32 + hi*16 + e] (contiguous in d)
            const bf16_t* kr0 = Kbh + (size_t)(k0 + l16) * DH + t2 * 32 + hi * 16;
            v8bf u0 = *(const v8bf*)(kr0);
            v8bf u1 = *(const v8bf*)(kr0 + 8);
            v16bf kf0;
#pragma unroll
            for (int i = 0; i < 8; ++i) { kf0[i] = u0[i]; kf0[8 + i] = u1[i]; }
            s0 = __builtin_amdgcn_wmma_f32_16x16x32_bf16(false, qa[t2], false, kf0,
                                                         (short)0, s0, false, false);
            const bf16_t* kr1 = Kbh + (size_t)(k0 + 16 + l16) * DH + t2 * 32 + hi * 16;
            v8bf w0 = *(const v8bf*)(kr1);
            v8bf w1 = *(const v8bf*)(kr1 + 8);
            v16bf kf1;
#pragma unroll
            for (int i = 0; i < 8; ++i) { kf1[i] = w0[i]; kf1[8 + i] = w1[i]; }
            s1 = __builtin_amdgcn_wmma_f32_16x16x32_bf16(false, qa[t2], false, kf1,
                                                         (short)0, s1, false, false);
        }

        // ---- causal mask + online softmax (C layout: row=hi*8+r, col=l16)
#pragma unroll
        for (int r = 0; r < 8; ++r) {
            const int row = q0 + hi * 8 + r;
            const int c0 = k0 + l16, c1 = c0 + 16;
            float v0 = (c0 <= row) ? s0[r] : -1e30f;
            float v1 = (c1 <= row) ? s1[r] : -1e30f;
            float mr = fmaxf(v0, v1);
#pragma unroll
            for (int off = 1; off < 16; off <<= 1)
                mr = fmaxf(mr, __shfl_xor(mr, off, 32));
            const float mnew = fmaxf(m[r], mr);
            const float corr = __expf(m[r] - mnew);
            const float p0 = __expf(v0 - mnew);
            const float p1 = __expf(v1 - mnew);
            float rs = p0 + p1;
#pragma unroll
            for (int off = 1; off < 16; off <<= 1)
                rs += __shfl_xor(rs, off, 32);
            lsum[r] = lsum[r] * corr + rs;
            m[r] = mnew;
#pragma unroll
            for (int dt = 0; dt < 4; ++dt) acc[dt][r] *= corr;
            P[(hi * 8 + r) * 32 + l16]      = (bf16_t)p0;
            P[(hi * 8 + r) * 32 + 16 + l16] = (bf16_t)p1;
        }

        // ---- P (C layout in LDS) -> A fragment
        const bf16_t* prow = P + l16 * 32;
        v8bf pr0 = *(const v8bf*)(prow + hi * 8);
        v8bf pr1 = *(const v8bf*)(prow + 16 + hi * 8);
        v16bf pa;
#pragma unroll
        for (int i = 0; i < 8; ++i) { pa[i] = pr0[i]; pa[8 + i] = pr1[i]; }

        // ---- O += P @ V : four 16x16 d-tiles
#pragma unroll
        for (int dt = 0; dt < 4; ++dt) {
            v16bf vb;
#pragma unroll
            for (int e = 0; e < 16; ++e)
                vb[e] = Vbh[(size_t)(k0 + hi * 8 + e + (e & 8)) * DH + dt * 16 + l16];
            acc[dt] = __builtin_amdgcn_wmma_f32_16x16x32_bf16(false, pa, false, vb,
                                                              (short)0, acc[dt],
                                                              false, false);
        }
    }

    // ---- normalize and write Y in [B,T,H*d] layout (ready for proj GEMM)
    const int b = bh >> 4, h = bh & 15;
#pragma unroll
    for (int r = 0; r < 8; ++r) {
        const float inv = 1.0f / lsum[r];
        const int t = q0 + hi * 8 + r;
#pragma unroll
        for (int dt = 0; dt < 4; ++dt)
            Yw[((size_t)(b * TT + t)) * DD + h * DH + dt * 16 + l16] =
                (bf16_t)(acc[dt][r] * inv);
    }
}

// ---------------------------------------------------------------------------
// Kernel 3: OUT = Y @ Wproj  (M=4096, N=1024, K=1024), f32 output.
// Same scheme as kernel 1: 16x64 wave tile, TDM-staged 32x64 W panel.
// ---------------------------------------------------------------------------
__global__ __launch_bounds__(256)
void proj_gemm_kernel(const bf16_t* __restrict__ y, const float* __restrict__ w,
                      float* __restrict__ out) {
    __shared__ float lds_wf[32 * 64];

    const int lane = threadIdx.x & 31;
    const int wv   = threadIdx.x >> 5;
    const int l16  = lane & 15;
    const int hi   = lane >> 4;
    const int ntg  = blockIdx.x;              // 0..15
    const int mt   = blockIdx.y * 8 + wv;     // 0..255

    const bf16_t* yrow = y + (size_t)(mt * 16 + l16) * DD;
    const unsigned lds_off = lds_addr_of(lds_wf);

    v8f c[4] = {};
    for (int kk = 0; kk < DD; kk += 32) {
        __syncthreads();
        if (wv == 0) {
            tdm_load_tile_f32(lds_off, w + (size_t)kk * DD + ntg * 64,
                              DD, DD, 64, 32, DD);
            __builtin_amdgcn_s_wait_tensorcnt(0);
        }
        if (kk + 32 < DD) {
            int kloc = threadIdx.x >> 3, n = (threadIdx.x & 7) * 8;
            __builtin_prefetch(&w[(size_t)(kk + 32 + kloc) * DD + ntg * 64 + n], 0, 1);
        }
        __syncthreads();

        v8bf r0 = *(const v8bf*)(yrow + kk + hi * 8);
        v8bf r1 = *(const v8bf*)(yrow + kk + 16 + hi * 8);
        v16bf a;
#pragma unroll
        for (int i = 0; i < 8; ++i) { a[i] = r0[i]; a[8 + i] = r1[i]; }

#pragma unroll
        for (int nt4 = 0; nt4 < 4; ++nt4) {
            v16bf bfrag;
#pragma unroll
            for (int e = 0; e < 16; ++e)
                bfrag[e] = (bf16_t)lds_wf[(hi * 8 + e + (e & 8)) * 64 + nt4 * 16 + l16];
            c[nt4] = __builtin_amdgcn_wmma_f32_16x16x32_bf16(false, a, false, bfrag,
                                                             (short)0, c[nt4],
                                                             false, false);
        }
    }

#pragma unroll
    for (int r = 0; r < 8; ++r) {
        float* orow = out + (size_t)(mt * 16 + hi * 8 + r) * DD + ntg * 64;
#pragma unroll
        for (int nt4 = 0; nt4 < 4; ++nt4)
            orow[nt4 * 16 + l16] = c[nt4][r];
    }
}

// ---------------------------------------------------------------------------
extern "C" void kernel_launch(void* const* d_in, const int* in_sizes, int n_in,
                              void* d_out, int out_size, void* d_ws, size_t ws_size,
                              hipStream_t stream) {
    const float* x      = (const float*)d_in[0];   // [B,T,D]   f32
    const float* w_qkv  = (const float*)d_in[1];   // [D,3D]    f32
    const float* w_proj = (const float*)d_in[2];   // [D,D]     f32
    float* out = (float*)d_out;                    // [B,T,D]   f32

    const size_t qkv_elems = (size_t)BB * NH * TT * DH;   // 4 Mi each
    bf16_t* Qw = (bf16_t*)d_ws;
    bf16_t* Kw = Qw + qkv_elems;
    bf16_t* Vw = Kw + qkv_elems;
    bf16_t* Yw = Vw + qkv_elems;                          // [B,T,D] bf16

    // 1) QKV GEMM: N groups = 3072/64 = 48, M wave-tile groups = 4096/128 = 32
    qkv_gemm_kernel<<<dim3(48, 32), 256, 0, stream>>>(x, w_qkv, Qw, Kw, Vw);

    // 2) Flash attention: 32 (b,h) * 128 q-tiles = 4096 waves = 512 blocks
    attn_kernel<<<dim3(512), 256, 0, stream>>>(Qw, Kw, Vw, Yw);

    // 3) Output projection: N groups = 1024/64 = 16, M groups = 32
    proj_gemm_kernel<<<dim3(16, 32), 256, 0, stream>>>(Yw, w_proj, out);
}
